// Trm_63282048139796
// MI455X (gfx1250) — matrix operand order repaired and checked by reading, not compiled, not run on hardware
//
#include <hip/hip_runtime.h>
#include <hip/hip_bf16.h>

typedef _Float16 half_t;
typedef __attribute__((ext_vector_type(16))) _Float16 v16h;
typedef __attribute__((ext_vector_type(8)))  _Float16 v8h;
typedef __attribute__((ext_vector_type(8)))  float    v8f;

#define BATCH 64
#define SEQ   1024
#define DMODEL 200
#define DP    224          // padded D: 14 tiles of 16, 7 K-steps of 32
#define NCLS  31
#define LN_CHUNKS 100      // 204800 / 2048
#define GWAVES 4

enum { EPI_RELU = 1, EPI_RES = 2, EPI_OUT32 = 4, EPI_OUT16 = 8, EPI_OUT16T = 16 };

// ---------------------------------------------------------------- weight prep
// Transpose weights to N-major f16 with zero padding to DP x DP.
// Wm (1600x200) is folded: cat = tile(c,8)  =>  cat@Wm == c @ sum_h Wm_h.
__global__ void prep_weights(const float* __restrict__ Wp, const float* __restrict__ Wm,
                             const float* __restrict__ W1, const float* __restrict__ W2,
                             const float* __restrict__ Wd,
                             half_t* WpT, half_t* WmT, half_t* W1T, half_t* W2T, half_t* WdT)
{
    int i = blockIdx.x * blockDim.x + threadIdx.x;
    if (i >= DP * DP) return;
    int n = i / DP, k = i % DP;
    bool in = (n < DMODEL) && (k < DMODEL);
    WpT[i] = in ? (half_t)Wp[k * DMODEL + n] : (half_t)0.0f;
    W1T[i] = in ? (half_t)W1[k * DMODEL + n] : (half_t)0.0f;
    W2T[i] = in ? (half_t)W2[k * DMODEL + n] : (half_t)0.0f;
    float s = 0.0f;
    if (in) {
#pragma unroll
        for (int h = 0; h < 8; ++h) s += Wm[(h * DMODEL + k) * DMODEL + n];
    }
    WmT[i] = (half_t)s;
    if (n < 32) {
        float wd = (n < NCLS && k < DMODEL) ? Wd[k * NCLS + n] : 0.0f;
        WdT[n * DP + k] = (half_t)wd;
    }
}

// ---------------------------------------------------------------- embedding
__global__ void embed_kernel(const int* __restrict__ x, const float* __restrict__ emb,
                             const float* __restrict__ pos, float* __restrict__ h32,
                             half_t* __restrict__ h16)
{
    long long i = (long long)blockIdx.x * blockDim.x + threadIdx.x;  // over B*S*DP
    int d = (int)(i % DP);
    long long bs = i / DP;
    int s = (int)(bs & (SEQ - 1));
    float v = 0.0f;
    if (d < DMODEL) {
        int tok = x[bs];
        v = emb[(long long)tok * DMODEL + d] + pos[s * DMODEL + d];
        h32[bs * DMODEL + d] = v;
    }
    h16[bs * DP + d] = (half_t)v;
}

// ---------------------------------------------------------------- generic WMMA GEMM
// C[M,N] = A[M,DP](f16, row-major, lda=DP) x W[DP,N] given as WT[N,DP] (f16, N-major)
// Epilogue: +bias, optional relu, optional +res32, writes f32 and/or padded f16 and/or
// per-batch-transposed f16 (layout [B][DP][SEQ], valid only when M == B*SEQ).
__global__ void gemm_wmma(const half_t* __restrict__ A, const half_t* __restrict__ WT,
                          const float* __restrict__ bias,
                          float* __restrict__ out32, int ldc,
                          const float* __restrict__ res32, int ldres,
                          half_t* __restrict__ out16, half_t* __restrict__ out16T,
                          int M, int Nvalid, int flags)
{
    int lane = threadIdx.x & 31;
    int wave = threadIdx.x >> 5;
    int mtile = blockIdx.x * GWAVES + wave;
    int m0 = mtile * 16;
    int n0 = blockIdx.y * 16;
    if (m0 >= M) return;

    int sel = lane >> 4;          // lane-half selects 16-wide K sub-chunk (contiguous k order, same in A and B)
    int r   = lane & 15;
    const half_t* arow = A  + (long long)(m0 + r) * DP + sel * 16;   // A operand rows
    const half_t* brow = WT + (long long)(n0 + r) * DP + sel * 16;   // B operand cols

    v8f acc = {};
#pragma unroll
    for (int k = 0; k < DP; k += 32) {
        __builtin_prefetch(arow + k + 64, 0, 0);
        __builtin_prefetch(brow + k + 64, 0, 0);
        v8h alo = *(const v8h*)(arow + k);
        v8h ahi = *(const v8h*)(arow + k + 8);
        v16h av;
#pragma unroll
        for (int t = 0; t < 8; ++t) { av[t] = alo[t]; av[t + 8] = ahi[t]; }
        v16h bv = *(const v16h*)(brow + k);
        acc = __builtin_amdgcn_wmma_f32_16x16x32_f16(false, av, false, bv,
                                                     (short)0, acc, false, false);
    }

    int n = n0 + (lane & 15);
    int mb = m0 + ((lane >> 4) ? 8 : 0);
    float bsv = (n < Nvalid) ? bias[n] : 0.0f;
#pragma unroll
    for (int v = 0; v < 8; ++v) {
        int m = mb + v;
        float val = acc[v] + bsv;
        if (flags & EPI_RELU) val = fmaxf(val, 0.0f);
        if ((flags & EPI_RES) && n < Nvalid) val += res32[(long long)m * ldres + n];
        if ((flags & EPI_OUT32) && n < Nvalid) out32[(long long)m * ldc + n] = val;
        half_t hv = (half_t)((n < Nvalid) ? val : 0.0f);
        if (flags & EPI_OUT16)  out16[(long long)m * DP + n] = hv;
        if (flags & EPI_OUT16T) out16T[((long long)(m >> 10) * DP + n) * SEQ + (m & (SEQ - 1))] = hv;
    }
}

// ---------------------------------------------------------------- fused attention
// One wave per (batch, 16-row query tile): scores = q k^T / sqrt(200) (WMMA, f32 in LDS),
// row softmax (v_exp_f32), probs f16 in LDS, then c = P V via WMMA with A from LDS.
__global__ void attn_kernel(const half_t* __restrict__ p16, const half_t* __restrict__ pT16,
                            half_t* __restrict__ c16)
{
    extern __shared__ __align__(16) char smem[];
    float*  sc = (float*)smem;                       // [16][SEQ] scores (64 KB)
    half_t* pr = (half_t*)(smem + 16 * SEQ * 4);     // [16][SEQ] probs  (32 KB)
    __shared__ float red[64];

    int b  = blockIdx.x;
    int q0 = blockIdx.y * 16;
    int lane = threadIdx.x;
    int sel = lane >> 4, r = lane & 15;

    const half_t* pb  = p16  + (long long)b * SEQ * DP;
    const half_t* ptb = pT16 + (long long)b * DP * SEQ;
    const half_t* qrow = pb + (long long)(q0 + r) * DP + sel * 16;
    const float rscale = 0.07071067811865475f;       // 1/sqrt(200)

    // ---- scores ----
    for (int kt = 0; kt < SEQ / 16; ++kt) {
        const half_t* krow = pb + (long long)(kt * 16 + r) * DP + sel * 16;  // B: col = key row
        v8f acc = {};
#pragma unroll
        for (int k = 0; k < DP; k += 32) {
            v8h alo = *(const v8h*)(qrow + k);
            v8h ahi = *(const v8h*)(qrow + k + 8);
            v16h av;
#pragma unroll
            for (int t = 0; t < 8; ++t) { av[t] = alo[t]; av[t + 8] = ahi[t]; }
            v16h bv = *(const v16h*)(krow + k);
            acc = __builtin_amdgcn_wmma_f32_16x16x32_f16(false, av, false, bv,
                                                         (short)0, acc, false, false);
        }
        int col = kt * 16 + (lane & 15);
        int rbase = (lane >> 4) * 8;
#pragma unroll
        for (int v = 0; v < 8; ++v) sc[(rbase + v) * SEQ + col] = acc[v] * rscale;
    }
    __syncthreads();

    // ---- softmax: lane handles row = lane&15, half-row segment = lane>>4 ----
    {
        int row = lane & 15, seg = lane >> 4;
        float* srow = sc + row * SEQ + seg * (SEQ / 2);
        float mx = -1e30f;
        for (int i = 0; i < SEQ / 2; ++i) mx = fmaxf(mx, srow[i]);
        red[lane] = mx;
        __syncthreads();
        float rowmax = fmaxf(red[row], red[row + 16]);
        float sum = 0.0f;
        for (int i = 0; i < SEQ / 2; ++i) {
            float e = __expf(srow[i] - rowmax);
            sum += e;
            srow[i] = e;
        }
        red[32 + lane] = sum;
        __syncthreads();
        float inv = 1.0f / (red[32 + row] + red[32 + row + 16]);
        half_t* prow = pr + row * SEQ + seg * (SEQ / 2);
        for (int i = 0; i < SEQ / 2; ++i) prow[i] = (half_t)(srow[i] * inv);
    }
    __syncthreads();

    // ---- c = P @ V  (V columns are contiguous rows of pT) ----
    const half_t* arow = pr + (long long)(lane & 15) * SEQ + sel * 16;  // A from LDS
    for (int dt = 0; dt < DP / 16; ++dt) {
        const half_t* vrow = ptb + (long long)(dt * 16 + (lane & 15)) * SEQ + sel * 16;
        v8f acc = {};
        for (int kc = 0; kc < SEQ; kc += 32) {
            v8h alo = *(const v8h*)(arow + kc);
            v8h ahi = *(const v8h*)(arow + kc + 8);
            v16h av;
#pragma unroll
            for (int t = 0; t < 8; ++t) { av[t] = alo[t]; av[t + 8] = ahi[t]; }
            v16h bv = *(const v16h*)(vrow + kc);
            acc = __builtin_amdgcn_wmma_f32_16x16x32_f16(false, av, false, bv,
                                                         (short)0, acc, false, false);
        }
        int n = dt * 16 + (lane & 15);
        int mb = q0 + ((lane >> 4) ? 8 : 0);
#pragma unroll
        for (int v = 0; v < 8; ++v)
            c16[((long long)b * SEQ + mb + v) * DP + n] = (half_t)acc[v];
    }
}

// ---------------------------------------------------------------- LayerNorm over (S,D) per batch
__global__ void ln_partial(const float* __restrict__ X, float* __restrict__ partials)
{
    int b = blockIdx.x, chunk = blockIdx.y;
    const float* xb = X + (long long)b * (SEQ * DMODEL) + chunk * 2048;
    float s = 0.0f, s2 = 0.0f;
    for (int i = threadIdx.x; i < 2048; i += 256) { float v = xb[i]; s += v; s2 += v * v; }
    __shared__ float sh0[256], sh1[256];
    sh0[threadIdx.x] = s; sh1[threadIdx.x] = s2;
    __syncthreads();
    for (int off = 128; off > 0; off >>= 1) {
        if (threadIdx.x < off) {
            sh0[threadIdx.x] += sh0[threadIdx.x + off];
            sh1[threadIdx.x] += sh1[threadIdx.x + off];
        }
        __syncthreads();
    }
    if (threadIdx.x == 0) {
        partials[(b * LN_CHUNKS + chunk) * 2 + 0] = sh0[0];
        partials[(b * LN_CHUNKS + chunk) * 2 + 1] = sh1[0];
    }
}

__global__ void ln_stats(const float* __restrict__ partials, float* __restrict__ stats)
{
    int b = threadIdx.x;
    if (b >= BATCH) return;
    float s = 0.0f, s2 = 0.0f;
    for (int c = 0; c < LN_CHUNKS; ++c) {
        s  += partials[(b * LN_CHUNKS + c) * 2 + 0];
        s2 += partials[(b * LN_CHUNKS + c) * 2 + 1];
    }
    const float invN = 1.0f / (SEQ * DMODEL);
    float mean = s * invN;
    float var  = s2 * invN - mean * mean;
    stats[b * 2 + 0] = mean;
    stats[b * 2 + 1] = rsqrtf(var + 1e-5f);
}

__global__ void ln_norm(const float* __restrict__ X, const float* __restrict__ stats,
                        float* __restrict__ h32, half_t* __restrict__ h16)
{
    long long i = (long long)blockIdx.x * blockDim.x + threadIdx.x;  // over B*S*DP
    int d = (int)(i % DP);
    long long bs = i / DP;
    int b = (int)(bs >> 10);
    float v = 0.0f;
    if (d < DMODEL) {
        v = (X[bs * DMODEL + d] - stats[b * 2]) * stats[b * 2 + 1];
        h32[bs * DMODEL + d] = v;
    }
    h16[bs * DP + d] = (half_t)v;
}

// ---------------------------------------------------------------- gather last seq position
__global__ void gather_last(const half_t* __restrict__ h16, half_t* __restrict__ last16)
{
    int i = blockIdx.x * blockDim.x + threadIdx.x;
    if (i >= BATCH * DP) return;
    int b = i / DP, d = i % DP;
    last16[i] = h16[((long long)b * SEQ + (SEQ - 1)) * DP + d];
}

// ---------------------------------------------------------------- host orchestration
extern "C" void kernel_launch(void* const* d_in, const int* in_sizes, int n_in,
                              void* d_out, int out_size, void* d_ws, size_t ws_size,
                              hipStream_t stream)
{
    const int*   x    = (const int*)  d_in[0];
    const float* emb  = (const float*)d_in[1];
    const float* pos  = (const float*)d_in[2];
    const float* Wp   = (const float*)d_in[3];
    const float* bp   = (const float*)d_in[4];
    const float* Wm   = (const float*)d_in[5];
    const float* bm   = (const float*)d_in[6];
    const float* W1   = (const float*)d_in[7];
    const float* b1   = (const float*)d_in[8];
    const float* W2   = (const float*)d_in[9];
    const float* b2   = (const float*)d_in[10];
    const float* Wd   = (const float*)d_in[11];
    const float* bd   = (const float*)d_in[12];
    float* out = (float*)d_out;

    // carve workspace
    char* ws = (char*)d_ws;
    size_t off = 0;
    auto carve = [&](size_t bytes) -> void* {
        void* p = ws + off;
        off += (bytes + 255) & ~(size_t)255;
        return p;
    };
    const size_t M = (size_t)BATCH * SEQ;
    float*  h32  = (float*) carve(M * DMODEL * 4);
    float*  t32  = (float*) carve(M * DMODEL * 4);
    half_t* h16  = (half_t*)carve(M * DP * 2);
    half_t* p16  = (half_t*)carve(M * DP * 2);
    half_t* pT16 = (half_t*)carve(M * DP * 2);
    half_t* c16  = (half_t*)carve(M * DP * 2);
    half_t* u16  = p16;  // reuse: p16 is dead after attention
    half_t* WpT  = (half_t*)carve(DP * DP * 2);
    half_t* WmT  = (half_t*)carve(DP * DP * 2);
    half_t* W1T  = (half_t*)carve(DP * DP * 2);
    half_t* W2T  = (half_t*)carve(DP * DP * 2);
    half_t* WdT  = (half_t*)carve(32 * DP * 2);
    float*  part = (float*) carve(BATCH * LN_CHUNKS * 2 * 4);
    float*  stat = (float*) carve(BATCH * 2 * 4);
    half_t* last = (half_t*)carve(BATCH * DP * 2);

    dim3 gemm_grid((unsigned)(M / 16 / GWAVES), DP / 16);     // (1024, 14)
    const int gemm_blk = 32 * GWAVES;
    const int elem_grid = (int)(M * DP / 256);                // 57344

    prep_weights<<<(DP * DP + 255) / 256, 256, 0, stream>>>(Wp, Wm, W1, W2, Wd,
                                                            WpT, WmT, W1T, W2T, WdT);
    embed_kernel<<<elem_grid, 256, 0, stream>>>(x, emb, pos, h32, h16);

    for (int it = 0; it < 2; ++it) {
        // p = h @ Wp + bp  -> f16 padded + f16 transposed
        gemm_wmma<<<gemm_grid, gemm_blk, 0, stream>>>(h16, WpT, bp, nullptr, 0, nullptr, 0,
                                                      p16, pT16, (int)M, DMODEL,
                                                      EPI_OUT16 | EPI_OUT16T);
        // c = softmax(p p^T / sqrt(200)) p
        attn_kernel<<<dim3(BATCH, SEQ / 16), 32, 16 * SEQ * 4 + 16 * SEQ * 2, stream>>>(p16, pT16, c16);
        // t = c @ Wm_eff + bm + h
        gemm_wmma<<<gemm_grid, gemm_blk, 0, stream>>>(c16, WmT, bm, t32, DMODEL, h32, DMODEL,
                                                      nullptr, nullptr, (int)M, DMODEL,
                                                      EPI_RES | EPI_OUT32);
        ln_partial<<<dim3(BATCH, LN_CHUNKS), 256, 0, stream>>>(t32, part);
        ln_stats<<<1, BATCH, 0, stream>>>(part, stat);
        ln_norm<<<elem_grid, 256, 0, stream>>>(t32, stat, h32, h16);
        // u = relu(h @ W1 + b1)
        gemm_wmma<<<gemm_grid, gemm_blk, 0, stream>>>(h16, W1T, b1, nullptr, 0, nullptr, 0,
                                                      u16, nullptr, (int)M, DMODEL,
                                                      EPI_RELU | EPI_OUT16);
        // t = u @ W2 + b2 + h
        gemm_wmma<<<gemm_grid, gemm_blk, 0, stream>>>(u16, W2T, b2, t32, DMODEL, h32, DMODEL,
                                                      nullptr, nullptr, (int)M, DMODEL,
                                                      EPI_RES | EPI_OUT32);
        ln_partial<<<dim3(BATCH, LN_CHUNKS), 256, 0, stream>>>(t32, part);
        ln_stats<<<1, BATCH, 0, stream>>>(part, stat);
        ln_norm<<<elem_grid, 256, 0, stream>>>(t32, stat, h32, h16);
    }

    // out = h[:, -1, :] @ Wd + bd   -> [64, 31]
    gather_last<<<(BATCH * DP + 255) / 256, 256, 0, stream>>>(h16, last);
    gemm_wmma<<<dim3(1, 2), gemm_blk, 0, stream>>>(last, WdT, bd, out, NCLS, nullptr, 0,
                                                   nullptr, nullptr, BATCH, NCLS, EPI_OUT32);
}
